// X_EfficientAdditiveAttnetion_14113262535050
// MI455X (gfx1250) — matrix operand order repaired and checked by reading, not compiled, not run on hardware
//
#include <hip/hip_runtime.h>

typedef __attribute__((ext_vector_type(16))) __bf16          v16bf;
typedef __attribute__((ext_vector_type(16))) unsigned short  v16us;
typedef __attribute__((ext_vector_type(8)))  float           v8f;

__device__ __forceinline__ unsigned short f2bf(float f) {
    unsigned int u = __float_as_uint(f);
    u += 0x7FFFu + ((u >> 16) & 1u);      // round-to-nearest-even
    return (unsigned short)(u >> 16);
}

// ---------------------------------------------------------------------------
// Pack f32 weight matrix W[K x Nout] (row-major) into bf16 B-fragment layout:
// per 32x16 tile (kt, nt): halves[lane][h] with
//   lane = ((k%32)/8 % 2 ? 16 : 0) + n%16 ,  h = k%8 + 8*((k%32)/16)  (table order)
// ---------------------------------------------------------------------------
__global__ void pack_w_kernel(const float* __restrict__ W,
                              unsigned short* __restrict__ dst,
                              int K, int Nout)
{
    int tid = blockIdx.x * 256 + threadIdx.x;
    if (tid >= K * Nout) return;
    int k = tid / Nout, n = tid % Nout;
    int kt = k >> 5, kin = k & 31;
    int nt = n >> 4, nin = n & 15;
    int lane = (((kin >> 3) & 1) << 4) + nin;
    int h = (kin & 7) | ((kin >> 4) << 3);
    int nT = Nout >> 4;
    size_t idx = ((((size_t)kt * nT + nt) * 32) + lane) * 16 + h;
    dst[idx] = f2bf(W[tid]);
}

// bps[m][n] = sum_j bp[m][j][n]
__global__ void bpsum_kernel(const float* __restrict__ bp, float* __restrict__ bps)
{
    int tid = blockIdx.x * 256 + threadIdx.x;       // 0..1535
    int m = tid >> 9, n = tid & 511;
    bps[tid] = bp[(m * 3 + 0) * 512 + n] + bp[(m * 3 + 1) * 512 + n] +
               bp[(m * 3 + 2) * 512 + n];
}

// ---------------------------------------------------------------------------
// WMMA bf16 GEMM with software-pipelined (double-buffered) K loop.
// C[row, :Nout] = (opt addend) + bias + sum_k scale(row,k)*A_seg[row,k]*Wpk
// A rows have stride 512 floats; reduction dim may be segmented
// (seg = k >> kSegBits, offset segStride floats) to realize concat-K.
// Block: 256 thr = 8 waves; wave tile 16 x 64; block tile 128 x 64.
// USEG / USEADD compile-time: no exec-mask predication in the hot loop.
// ---------------------------------------------------------------------------
template <bool USEG>
__device__ __forceinline__ void load_frags(
    const float* __restrict__ A, size_t segStride, int kSegBits, int kmask,
    size_t rowOff, const float* __restrict__ grow,
    const unsigned short* __restrict__ Wpk, int nT, int nt0, int lane, int k8,
    int kk,
    float4& a0, float4& a1, float4& a2, float4& a3,
    float4& g0, float4& g1, float4& g2, float4& g3,
    v16bf& b0, v16bf& b1, v16bf& b2, v16bf& b3)
{
    const int kb = kk + k8;
    const int kloc = kb & kmask;
    const float* ap = A + (size_t)(kb >> kSegBits) * segStride + rowOff + kloc;
    a0 = ((const float4*)ap)[0];            // K = kb .. kb+3
    a1 = ((const float4*)ap)[1];            // K = kb+4 .. kb+7
    a2 = ((const float4*)(ap + 16))[0];     // K = kb+16 .. kb+19
    a3 = ((const float4*)(ap + 16))[1];     // K = kb+20 .. kb+23
    if (USEG) {
        const float* gp = grow + kloc;
        g0 = ((const float4*)gp)[0];
        g1 = ((const float4*)gp)[1];
        g2 = ((const float4*)(gp + 16))[0];
        g3 = ((const float4*)(gp + 16))[1];
    }
    const size_t bb = ((((size_t)(kk >> 5) * nT + nt0) * 32) + lane) * 16;
    b0 = *(const v16bf*)(Wpk + bb);
    b1 = *(const v16bf*)(Wpk + bb + 512);
    b2 = *(const v16bf*)(Wpk + bb + 1024);
    b3 = *(const v16bf*)(Wpk + bb + 1536);
}

template <bool USEG>
__device__ __forceinline__ void compute_frags(
    float4 a0, float4 a1, float4 a2, float4 a3,
    float4 g0, float4 g1, float4 g2, float4 g3,
    v16bf b0, v16bf b1, v16bf b2, v16bf b3,
    v8f& c0, v8f& c1, v8f& c2, v8f& c3)
{
    if (USEG) {
        a0.x *= g0.x; a0.y *= g0.y; a0.z *= g0.z; a0.w *= g0.w;
        a1.x *= g1.x; a1.y *= g1.y; a1.z *= g1.z; a1.w *= g1.w;
        a2.x *= g2.x; a2.y *= g2.y; a2.z *= g2.z; a2.w *= g2.w;
        a3.x *= g3.x; a3.y *= g3.y; a3.z *= g3.z; a3.w *= g3.w;
    }
    v16us au;
    au[0]  = f2bf(a0.x); au[1]  = f2bf(a0.y); au[2]  = f2bf(a0.z); au[3]  = f2bf(a0.w);
    au[4]  = f2bf(a1.x); au[5]  = f2bf(a1.y); au[6]  = f2bf(a1.z); au[7]  = f2bf(a1.w);
    au[8]  = f2bf(a2.x); au[9]  = f2bf(a2.y); au[10] = f2bf(a2.z); au[11] = f2bf(a2.w);
    au[12] = f2bf(a3.x); au[13] = f2bf(a3.y); au[14] = f2bf(a3.z); au[15] = f2bf(a3.w);
    v16bf af = __builtin_bit_cast(v16bf, au);
    c0 = __builtin_amdgcn_wmma_f32_16x16x32_bf16(false, af, false, b0, (short)0, c0, false, false);
    c1 = __builtin_amdgcn_wmma_f32_16x16x32_bf16(false, af, false, b1, (short)0, c1, false, false);
    c2 = __builtin_amdgcn_wmma_f32_16x16x32_bf16(false, af, false, b2, (short)0, c2, false, false);
    c3 = __builtin_amdgcn_wmma_f32_16x16x32_bf16(false, af, false, b3, (short)0, c3, false, false);
}

template <bool USEG, bool USEADD>
__global__ __launch_bounds__(256)
void gemm_bf16_kernel(const float* __restrict__ A, size_t segStride, int kSegBits,
                      const unsigned short* __restrict__ Wpk,
                      const float* __restrict__ bias,
                      const float* __restrict__ addend,
                      const float* __restrict__ scaleG,
                      float* __restrict__ C,
                      int Nout, int Ktot)
{
    const int lane  = threadIdx.x & 31;
    const int wv    = threadIdx.x >> 5;
    const int tileM = blockIdx.x * 128 + wv * 16;
    const int tileN = blockIdx.y * 64;
    const int nT    = Nout >> 4;
    const int nt0   = tileN >> 4;

    const int row   = tileM + (lane & 15);
    const int k8    = (lane >> 4) << 3;        // 0 or 8 (A layout: two 8-wide K chunks)
    const int kmask = (1 << kSegBits) - 1;
    const size_t rowOff = ((size_t)row << 9);
    const float* grow = USEG ? (scaleG + ((size_t)(row >> 10) << 9)) : nullptr;

    v8f c0 = {}, c1 = {}, c2 = {}, c3 = {};

    // ping-pong fragment buffers (raw f32 A + optional G, packed bf16 B)
    float4 xa0, xa1, xa2, xa3, ya0, ya1, ya2, ya3;
    float4 xg0 = {}, xg1 = {}, xg2 = {}, xg3 = {};
    float4 yg0 = {}, yg1 = {}, yg2 = {}, yg3 = {};
    v16bf  xb0, xb1, xb2, xb3, yb0, yb1, yb2, yb3;

    load_frags<USEG>(A, segStride, kSegBits, kmask, rowOff, grow, Wpk, nT, nt0, lane, k8,
                     0, xa0, xa1, xa2, xa3, xg0, xg1, xg2, xg3, xb0, xb1, xb2, xb3);

    int kk = 32;
    for (; kk + 32 < Ktot; kk += 64) {
        load_frags<USEG>(A, segStride, kSegBits, kmask, rowOff, grow, Wpk, nT, nt0, lane, k8,
                         kk, ya0, ya1, ya2, ya3, yg0, yg1, yg2, yg3, yb0, yb1, yb2, yb3);
        compute_frags<USEG>(xa0, xa1, xa2, xa3, xg0, xg1, xg2, xg3,
                            xb0, xb1, xb2, xb3, c0, c1, c2, c3);
        load_frags<USEG>(A, segStride, kSegBits, kmask, rowOff, grow, Wpk, nT, nt0, lane, k8,
                         kk + 32, xa0, xa1, xa2, xa3, xg0, xg1, xg2, xg3, xb0, xb1, xb2, xb3);
        compute_frags<USEG>(ya0, ya1, ya2, ya3, yg0, yg1, yg2, yg3,
                            yb0, yb1, yb2, yb3, c0, c1, c2, c3);
    }
    // tail: chunk kk (== Ktot-32) still unloaded; X holds chunk kk-32
    load_frags<USEG>(A, segStride, kSegBits, kmask, rowOff, grow, Wpk, nT, nt0, lane, k8,
                     kk, ya0, ya1, ya2, ya3, yg0, yg1, yg2, yg3, yb0, yb1, yb2, yb3);
    compute_frags<USEG>(xa0, xa1, xa2, xa3, xg0, xg1, xg2, xg3,
                        xb0, xb1, xb2, xb3, c0, c1, c2, c3);
    compute_frags<USEG>(ya0, ya1, ya2, ya3, yg0, yg1, yg2, yg3,
                        yb0, yb1, yb2, yb3, c0, c1, c2, c3);

    // Epilogue. C/D layout: VGPR i, lanes 0-15 -> M=i, lanes 16-31 -> M=8+i; N = lane%16.
    v8f cc[4] = {c0, c1, c2, c3};
    const int erow = tileM + ((lane >> 4) << 3);
    const int ecol = tileN + (lane & 15);
#pragma unroll
    for (int j = 0; j < 4; ++j) {
        const int col = ecol + j * 16;
        const float b = bias[col];
#pragma unroll
        for (int i = 0; i < 8; ++i) {
            size_t idx = (size_t)(erow + i) * Nout + col;
            float v = cc[j][i] + b;
            if (USEADD) v += addend[idx];
            C[idx] = v;
        }
    }
}

// ---------------------------------------------------------------------------
// Row-wise L2 normalize (rows of 512 f32). One wave32 per row, 8 rows/block.
// ---------------------------------------------------------------------------
__global__ __launch_bounds__(256)
void l2norm_rows_kernel(float* __restrict__ X)
{
    const int row  = blockIdx.x * 8 + (threadIdx.x >> 5);
    const int lane = threadIdx.x & 31;
    float* p = X + ((size_t)row << 9) + lane * 16;
    float4 v0 = ((float4*)p)[0], v1 = ((float4*)p)[1],
           v2 = ((float4*)p)[2], v3 = ((float4*)p)[3];
    float s = v0.x*v0.x + v0.y*v0.y + v0.z*v0.z + v0.w*v0.w
            + v1.x*v1.x + v1.y*v1.y + v1.z*v1.z + v1.w*v1.w
            + v2.x*v2.x + v2.y*v2.y + v2.z*v2.z + v2.w*v2.w
            + v3.x*v3.x + v3.y*v3.y + v3.z*v3.z + v3.w*v3.w;
#pragma unroll
    for (int off = 16; off; off >>= 1) s += __shfl_xor(s, off, 32);
    const float inv = 1.0f / fmaxf(sqrtf(s), 1e-12f);
    v0.x*=inv; v0.y*=inv; v0.z*=inv; v0.w*=inv;
    v1.x*=inv; v1.y*=inv; v1.z*=inv; v1.w*=inv;
    v2.x*=inv; v2.y*=inv; v2.z*=inv; v2.w*=inv;
    v3.x*=inv; v3.y*=inv; v3.z*=inv; v3.w*=inv;
    ((float4*)p)[0] = v0; ((float4*)p)[1] = v1;
    ((float4*)p)[2] = v2; ((float4*)p)[3] = v3;
}

// t[row] = SCALE * dot(Q[row,:], wg[m,:])   row = m*16384 + b*1024 + n
__global__ __launch_bounds__(256)
void dotg_kernel(const float* __restrict__ Q, const float* __restrict__ wg,
                 float* __restrict__ t)
{
    const int row  = blockIdx.x * 8 + (threadIdx.x >> 5);
    const int lane = threadIdx.x & 31;
    const int m    = row >> 14;
    const float* q = Q  + ((size_t)row << 9) + lane * 16;
    const float* w = wg + m * 512 + lane * 16;
    float s = 0.f;
#pragma unroll
    for (int i = 0; i < 16; ++i) s += q[i] * w[i];
#pragma unroll
    for (int off = 16; off; off >>= 1) s += __shfl_xor(s, off, 32);
    if (lane == 0) t[row] = s * 0.0625f;   // SCALE = 256^-0.5
}

// Per (m,b): a = t / max(||t||_2, eps) over n;  G[d] = sum_n a[n]*Q[m,b,n,d]
__global__ __launch_bounds__(256)
void gvec_kernel(const float* __restrict__ Q, const float* __restrict__ t,
                 float* __restrict__ G)
{
    __shared__ float tl[1024];
    __shared__ float red[8];
    const int mb = blockIdx.x;                 // m*16 + b, 0..47
    const float* tp = t + (size_t)mb * 1024;
    float ss = 0.f;
    for (int i = threadIdx.x; i < 1024; i += 256) {
        float v = tp[i]; tl[i] = v; ss += v * v;
    }
#pragma unroll
    for (int off = 16; off; off >>= 1) ss += __shfl_xor(ss, off, 32);
    if ((threadIdx.x & 31) == 0) red[threadIdx.x >> 5] = ss;
    __syncthreads();
    float tot = 0.f;
#pragma unroll
    for (int w = 0; w < 8; ++w) tot += red[w];
    const float inv = 1.0f / fmaxf(sqrtf(tot), 1e-12f);

    const float* qb = Q + (size_t)mb * 1024 * 512;
    const int d0 = threadIdx.x, d1 = threadIdx.x + 256;
    float acc0 = 0.f, acc1 = 0.f;
    for (int n = 0; n < 1024; ++n) {
        const float a = tl[n] * inv;
        acc0 += a * qb[((size_t)n << 9) + d0];
        acc1 += a * qb[((size_t)n << 9) + d1];
    }
    G[((size_t)mb << 9) + d0] = acc0;
    G[((size_t)mb << 9) + d1] = acc1;
}

// ---------------------------------------------------------------------------
extern "C" void kernel_launch(void* const* d_in, const int* in_sizes, int n_in,
                              void* d_out, int out_size, void* d_ws, size_t ws_size,
                              hipStream_t stream)
{
    (void)in_sizes; (void)n_in; (void)out_size; (void)ws_size;
    const float* x[3] = {(const float*)d_in[0], (const float*)d_in[1], (const float*)d_in[2]};
    const float* Wq = (const float*)d_in[3];
    const float* bq = (const float*)d_in[4];
    const float* Wk = (const float*)d_in[5];
    const float* bk = (const float*)d_in[6];
    const float* wg = (const float*)d_in[7];
    const float* Wp = (const float*)d_in[8];
    const float* bp = (const float*)d_in[9];
    const float* Wf = (const float*)d_in[10];
    const float* bf = (const float*)d_in[11];
    float* out = (float*)d_out;

    const int Bn = 16, Dd = 512, TD = 256;
    const size_t MB   = (size_t)Bn * 1024;    // 16384 rows / branch
    const size_t matF = MB * Dd;              // floats per branch matrix

    float* Qbuf = (float*)d_ws;
    float* Kbuf = Qbuf + 3 * matF;
    float* accb = Kbuf + 3 * matF;
    float* tbuf = accb + 3 * matF;
    float* Gbuf = tbuf + 3 * MB;
    float* bps  = Gbuf + (size_t)3 * Bn * Dd;
    unsigned short* Wq_pk = (unsigned short*)(bps + 3 * Dd);
    unsigned short* Wk_pk = Wq_pk + (size_t)3 * Dd * Dd;
    unsigned short* Wp_pk = Wk_pk + (size_t)3 * Dd * Dd;
    unsigned short* Wf_pk = Wp_pk + (size_t)3 * 3 * Dd * Dd;

    // ---- pack weights to bf16 fragment layout ----
    for (int m = 0; m < 3; ++m) {
        pack_w_kernel<<<dim3((Dd * Dd + 255) / 256), 256, 0, stream>>>(
            Wq + (size_t)m * Dd * Dd, Wq_pk + (size_t)m * Dd * Dd, Dd, Dd);
        pack_w_kernel<<<dim3((Dd * Dd + 255) / 256), 256, 0, stream>>>(
            Wk + (size_t)m * Dd * Dd, Wk_pk + (size_t)m * Dd * Dd, Dd, Dd);
        pack_w_kernel<<<dim3((3 * Dd * Dd + 255) / 256), 256, 0, stream>>>(
            Wp + (size_t)m * 3 * Dd * Dd, Wp_pk + (size_t)m * 3 * Dd * Dd, 3 * Dd, Dd);
        pack_w_kernel<<<dim3((Dd * TD + 255) / 256), 256, 0, stream>>>(
            Wf + (size_t)m * Dd * TD, Wf_pk + (size_t)m * Dd * TD, Dd, TD);
    }
    bpsum_kernel<<<dim3(6), 256, 0, stream>>>(bp, bps);

    // ---- stage 1: Q/K projections ----
    for (int m = 0; m < 3; ++m) {
        dim3 g(MB / 128, Dd / 64);
        gemm_bf16_kernel<false, false><<<g, 256, 0, stream>>>(
            x[m], 0, 30, Wq_pk + (size_t)m * Dd * Dd, bq + m * Dd,
            nullptr, nullptr, Qbuf + m * matF, Dd, Dd);
        gemm_bf16_kernel<false, false><<<g, 256, 0, stream>>>(
            x[m], 0, 30, Wk_pk + (size_t)m * Dd * Dd, bk + m * Dd,
            nullptr, nullptr, Kbuf + m * matF, Dd, Dd);
    }

    // ---- L2 normalize all Q and K rows (contiguous buffers) ----
    l2norm_rows_kernel<<<dim3((unsigned)(6 * MB / 8)), 256, 0, stream>>>(Qbuf);

    // ---- attention vector t, then G ----
    dotg_kernel<<<dim3((unsigned)(3 * MB / 8)), 256, 0, stream>>>(Qbuf, wg, tbuf);
    gvec_kernel<<<dim3(48), 256, 0, stream>>>(Qbuf, tbuf, Gbuf);

    // ---- Wp stage: acc[m] = Q[m] + [G*K0|G*K1|G*K2] @ Wp[m] + sum_j bp[m,j]
    //      single K=1536 GEMM: seg = k>>9 walks the three K buffers ----
    for (int m = 0; m < 3; ++m) {
        dim3 g(MB / 128, Dd / 64);
        gemm_bf16_kernel<true, true><<<g, 256, 0, stream>>>(
            Kbuf, matF, 9, Wp_pk + (size_t)m * 3 * Dd * Dd, bps + m * Dd,
            Qbuf + m * matF, Gbuf + (size_t)m * Bn * Dd,
            accb + m * matF, Dd, 3 * Dd);
    }

    // ---- Wf stage (reference bug: FIDX = {0,1,1}) ----
    const int FIDX[3] = {0, 1, 1};
    for (int m = 0; m < 3; ++m) {
        dim3 g(MB / 128, TD / 64);
        gemm_bf16_kernel<false, false><<<g, 256, 0, stream>>>(
            accb + m * matF, 0, 30, Wf_pk + (size_t)FIDX[m] * Dd * TD, bf + FIDX[m] * TD,
            nullptr, nullptr, out + (size_t)m * MB * TD, TD, Dd);
    }
}
